// QuantumSpectralBranch_5377299054629
// MI455X (gfx1250) — matrix operand order repaired and checked by reading, not compiled, not run on hardware
//
#include <hip/hip_runtime.h>
#include <math.h>

// MI455X / gfx1250: wave32, WMMA 16x16x32 f16 -> f32 accumulate.
typedef __attribute__((ext_vector_type(16))) _Float16 v16h;
typedef __attribute__((ext_vector_type(8)))  float    v8f;

#define NQ 8
#define DIM 256          // 2^NQ amplitudes
#define BATCH 32768
#define ROWS_PER_WG 64
#define WAVES 4
#define S_STRIDE 258     // 256 + 2 halfs pad -> kills LDS bank conflicts on A-frag loads
#define EX_STRIDE 66     // 64 + 2 halfs pad

// ---------------------------------------------------------------------------
// Kernel A: build the fixed 256x256 complex circuit unitary U from q_weights.
// Block c simulates U|e_c>; thread t holds amplitude t. Stores U^T as f16:
//   Uc[n][k] = Re(U[n][k]) for n<256 ; Uc[256+n][k] = Im(U[n][k])
// Row-major (k contiguous) so WMMA B-fragments are contiguous 16B loads.
// ---------------------------------------------------------------------------
__global__ __launch_bounds__(256) void build_unitary(const float* __restrict__ qw,
                                                     _Float16* __restrict__ Uc) {
  __shared__ float2 lds[DIM];
  const int t = threadIdx.x;
  const int c = blockIdx.x;
  float2 amp = make_float2(t == c ? 1.0f : 0.0f, 0.0f);

  for (int l = 0; l < 2; ++l) {
    // Rot(phi, theta, omega) = RZ(omega) RY(theta) RZ(phi) on each wire
    for (int i = 0; i < NQ; ++i) {
      const float phi   = qw[(l * NQ + i) * 3 + 0];
      const float theta = qw[(l * NQ + i) * 3 + 1];
      const float omega = qw[(l * NQ + i) * 3 + 2];
      const float ct = __cosf(0.5f * theta), st = __sinf(0.5f * theta);
      const float ap = 0.5f * (phi + omega), am = 0.5f * (phi - omega);
      const float2 ep = make_float2(__cosf(ap), -__sinf(ap)); // exp(-i(phi+omega)/2)
      const float2 em = make_float2(__cosf(am), -__sinf(am)); // exp(-i(phi-omega)/2)
      // U = [[ep*ct, -conj(em)*st], [em*st, conj(ep)*ct]]
      const float2 u00 = make_float2( ep.x * ct,  ep.y * ct);
      const float2 u01 = make_float2(-em.x * st,  em.y * st);
      const float2 u10 = make_float2( em.x * st,  em.y * st);
      const float2 u11 = make_float2( ep.x * ct, -ep.y * ct);
      const int mask = 1 << (7 - i);     // qubit i lives at bit (7-i)
      lds[t] = amp; __syncthreads();
      const float2 pa = lds[t ^ mask]; __syncthreads();
      if ((t & mask) == 0) {
        amp = make_float2(u00.x*amp.x - u00.y*amp.y + u01.x*pa.x - u01.y*pa.y,
                          u00.x*amp.y + u00.y*amp.x + u01.x*pa.y + u01.y*pa.x);
      } else {
        amp = make_float2(u10.x*pa.x - u10.y*pa.y + u11.x*amp.x - u11.y*amp.y,
                          u10.x*pa.y + u10.y*pa.x + u11.x*amp.y + u11.y*amp.x);
      }
    }
    const int r = (l % (NQ - 1)) + 1;    // ring range: layer0 -> 1, layer1 -> 2
    for (int i = 0; i < NQ; ++i) {       // CNOT(control=i, target=(i+r)%8)
      const int bc = 7 - i;
      const int bt = 7 - ((i + r) & 7);
      lds[t] = amp; __syncthreads();
      const int src = ((t >> bc) & 1) ? (t ^ (1 << bt)) : t;
      amp = lds[src]; __syncthreads();
    }
  }
  Uc[t * DIM + c]         = (_Float16)amp.x;   // real rows 0..255
  Uc[(t + DIM) * DIM + c] = (_Float16)amp.y;   // imag rows 256..511
}

// ---------------------------------------------------------------------------
// Kernel A2: fold the Z-sign matrix into W:  M2T[j][n] = sum_i sign(n,i)*W[i][j]
// sign(n,i) = +1 if bit(7-i) of n is 0 else -1.
// ---------------------------------------------------------------------------
__global__ __launch_bounds__(256) void build_m2(const float* __restrict__ W,
                                                _Float16* __restrict__ M2T) {
  const int j = blockIdx.x;   // 0..63
  const int n = threadIdx.x;  // 0..255
  float acc = 0.f;
#pragma unroll
  for (int i = 0; i < NQ; ++i) {
    const float w = W[i * 64 + j];
    acc += (((n >> (7 - i)) & 1) ? -w : w);
  }
  M2T[j * DIM + n] = (_Float16)acc;
}

// ---------------------------------------------------------------------------
// Fragment loaders (layouts per CDNA5 ISA §7.12.2).
// A 16x32 f16: lane L -> row M=L%16, h=L/16; reg j<4: K=2j+8h ; j>=4: K=16+2(j-4)+8h
// B 32x16 f16: lane L -> col N=L%16, h=L/16; reg j<4: K=2j+8h ; j>=4: K=16+2(j-4)+8h (same as A)
// ---------------------------------------------------------------------------
__device__ __forceinline__ v16h load_a_lds(const _Float16* base, int m, int h,
                                           int kbase, int stride) {
  union { v16h v; unsigned u[8]; } f;
  const unsigned* p = (const unsigned*)base;
  int off = m * stride + kbase + 8 * h;       // halfs; even by construction
#pragma unroll
  for (int j = 0; j < 4; ++j) f.u[j] = p[(off >> 1) + j];
  off += 16;
#pragma unroll
  for (int j = 0; j < 4; ++j) f.u[4 + j] = p[(off >> 1) + j];
  return f.v;
}

__device__ __forceinline__ const _Float16* b_addr(const _Float16* __restrict__ Bm,
                                                  int n0, int lane, int kbase) {
  const int n = n0 + (lane & 15);
  const int h = lane >> 4;
  return Bm + n * DIM + kbase + 8 * h;        // 16B-aligned fragment base
}

__device__ __forceinline__ v16h load_b_glb(const _Float16* __restrict__ Bm,
                                           int n0, int lane, int kbase) {
  union { v16h v; uint4 q[2]; } f;
  const _Float16* p = b_addr(Bm, n0, lane, kbase);
  f.q[0] = *(const uint4*)(p);
  f.q[1] = *(const uint4*)(p + 16);
  return f.v;
}

// ---------------------------------------------------------------------------
// Kernel B: fused  S-build -> GEMM1 (re|im) -> probs -> GEMM2 -> LN -> GELU
// 128 threads (4 waves), 64 batch rows per workgroup; wave w owns rows w*16..+15.
// B-fragments of each K-step are preloaded together (load clause + staggered
// waits) and the NEXT K-step is software-prefetched (global_prefetch_b8) so
// steady-state WMMAs hit in WGP$ instead of stalling on L2.
// ---------------------------------------------------------------------------
__global__ __launch_bounds__(128, 1) void qsb_main(const float* __restrict__ pca,
                                                   const _Float16* __restrict__ Uc,
                                                   const _Float16* __restrict__ M2T,
                                                   const float* __restrict__ bias,
                                                   const float* __restrict__ gam,
                                                   const float* __restrict__ bet,
                                                   float* __restrict__ out) {
  __shared__ _Float16 S[ROWS_PER_WG * S_STRIDE];        // 33 KB  A-tile (f16)
  __shared__ _Float16 EX[WAVES * 16 * EX_STRIDE];       // 8.3 KB probs exchange

  const int tid = threadIdx.x;

  // ---- phase 0: build S[row][k] = P_hi[k>>4] * P_lo[k&15] (real RY product state)
  {
    const int row = tid & 63;
    const int hi  = tid >> 6;                 // which half of k-space this thread fills
    const int gRow = blockIdx.x * ROWS_PER_WG + row;
    float f0[8], f1[8];                       // f*[b]: cos/sin of half-angle at bit b
#pragma unroll
    for (int b = 0; b < 8; ++b) {
      const float x   = pca[gRow * 8 + (7 - b)];
      const float ang = 3.14159265358979f / (1.0f + __expf(-x)); // sigmoid(x)*pi
      f0[b] = __cosf(0.5f * ang);
      f1[b] = __sinf(0.5f * ang);
    }
    float Plo[16];
#pragma unroll
    for (int k = 0; k < 16; ++k)
      Plo[k] = ((k & 1) ? f1[0] : f0[0]) * ((k & 2) ? f1[1] : f0[1]) *
               ((k & 4) ? f1[2] : f0[2]) * ((k & 8) ? f1[3] : f0[3]);
    const float g7 = hi ? f1[7] : f0[7];
    float Phi[8];
#pragma unroll
    for (int j = 0; j < 8; ++j)
      Phi[j] = ((j & 1) ? f1[4] : f0[4]) * ((j & 2) ? f1[5] : f0[5]) *
               ((j & 4) ? f1[6] : f0[6]) * g7;
    _Float16* srow = &S[row * S_STRIDE + hi * 128];
#pragma unroll
    for (int j = 0; j < 8; ++j)
#pragma unroll
      for (int k = 0; k < 16; ++k)
        srow[j * 16 + k] = (_Float16)(Phi[j] * Plo[k]);
  }
  __syncthreads();

  const int w    = tid >> 5;
  const int lane = tid & 31;
  const int m    = lane & 15;
  const int h    = lane >> 4;
  _Float16* ex = &EX[w * 16 * EX_STRIDE];
  const _Float16* srow0 = &S[(w * 16) * S_STRIDE];

  v8f acc2[4] = {};                       // 16x64 f32 pre-activation h (minus bias)

  for (int g = 0; g < 4; ++g) {           // 4 n-groups of 64 state columns
    v8f ar[4] = {}, ai[4] = {};
#pragma unroll
    for (int kt = 0; kt < 8; ++kt) {      // K = 256 in chunks of 32
      const v16h a = load_a_lds(srow0, m, h, kt * 32, S_STRIDE);
      // Preload all 8 B-fragments for this K-step -> one load clause, then MACs.
      v16h br[4], bi[4];
#pragma unroll
      for (int i = 0; i < 4; ++i) {
        br[i] = load_b_glb(Uc,       g * 64 + i * 16, lane, kt * 32);
        bi[i] = load_b_glb(Uc, 256 + g * 64 + i * 16, lane, kt * 32);
      }
      // Software-prefetch next K-step's fragments into WGP$ (global_prefetch_b8).
      if (kt < 7) {
#pragma unroll
        for (int i = 0; i < 4; ++i) {
          __builtin_prefetch(b_addr(Uc,       g * 64 + i * 16, lane, (kt + 1) * 32), 0, 3);
          __builtin_prefetch(b_addr(Uc, 256 + g * 64 + i * 16, lane, (kt + 1) * 32), 0, 3);
        }
      }
#pragma unroll
      for (int i = 0; i < 4; ++i) {
        ar[i] = __builtin_amdgcn_wmma_f32_16x16x32_f16(false, a, false, br[i],
                                                       (short)0, ar[i], false, false);
        ai[i] = __builtin_amdgcn_wmma_f32_16x16x32_f16(false, a, false, bi[i],
                                                       (short)0, ai[i], false, false);
      }
    }
    // probs = re^2 + im^2, re-fragment via LDS (C/D layout -> A layout)
#pragma unroll
    for (int i = 0; i < 4; ++i)
#pragma unroll
      for (int r = 0; r < 8; ++r) {
        const float p = ar[i][r] * ar[i][r] + ai[i][r] * ai[i][r];
        ex[(r + 8 * h) * EX_STRIDE + (m + 16 * i)] = (_Float16)p;
      }
    __syncthreads();                      // uniform: all waves at same g
    // Stage-2: preload both A2 fragments and all 8 B2 fragments, then 8 WMMAs.
    {
      v16h a2[2], b2[2][4];
#pragma unroll
      for (int c2 = 0; c2 < 2; ++c2) {
        a2[c2] = load_a_lds(ex, m, h, c2 * 32, EX_STRIDE);
        const int k2g = g * 64 + c2 * 32;
#pragma unroll
        for (int j2 = 0; j2 < 4; ++j2)
          b2[c2][j2] = load_b_glb(M2T, j2 * 16, lane, k2g);
      }
#pragma unroll
      for (int c2 = 0; c2 < 2; ++c2)
#pragma unroll
        for (int j2 = 0; j2 < 4; ++j2)
          acc2[j2] = __builtin_amdgcn_wmma_f32_16x16x32_f16(false, a2[c2], false, b2[c2][j2],
                                                            (short)0, acc2[j2], false, false);
    }
    __syncthreads();
  }

  // ---- epilogue: bias, LayerNorm over 64 cols (16-lane shfl reduce), exact GELU
  float bv[4], gv[4], bev[4];
#pragma unroll
  for (int j2 = 0; j2 < 4; ++j2) {
    const int n = m + 16 * j2;
    bv[j2] = bias[n]; gv[j2] = gam[n]; bev[j2] = bet[n];
  }
  const int rowBase = blockIdx.x * ROWS_PER_WG + w * 16;
#pragma unroll
  for (int r = 0; r < 8; ++r) {           // C layout: reg r -> row m = r + 8h
    float v[4];
    float sum = 0.f, sq = 0.f;
#pragma unroll
    for (int j2 = 0; j2 < 4; ++j2) {
      v[j2] = acc2[j2][r] + bv[j2];
      sum += v[j2]; sq += v[j2] * v[j2];
    }
#pragma unroll
    for (int d = 1; d < 16; d <<= 1) {    // reduce across the 16 lanes of this row
      sum += __shfl_xor(sum, d, 32);
      sq  += __shfl_xor(sq,  d, 32);
    }
    const float mean = sum * (1.0f / 64.0f);
    const float var  = sq * (1.0f / 64.0f) - mean * mean;
    const float rstd = rsqrtf(var + 1e-5f);
    const int gr = rowBase + r + 8 * h;
#pragma unroll
    for (int j2 = 0; j2 < 4; ++j2) {
      const float hn = (v[j2] - mean) * rstd * gv[j2] + bev[j2];
      out[gr * 64 + m + 16 * j2] = 0.5f * hn * (1.0f + erff(hn * 0.70710678118f));
    }
  }
}

// ---------------------------------------------------------------------------
extern "C" void kernel_launch(void* const* d_in, const int* in_sizes, int n_in,
                              void* d_out, int out_size, void* d_ws, size_t ws_size,
                              hipStream_t stream) {
  (void)in_sizes; (void)n_in; (void)out_size; (void)ws_size;
  const float* pca = (const float*)d_in[0];
  const float* qw  = (const float*)d_in[1];
  const float* W   = (const float*)d_in[2];
  const float* b   = (const float*)d_in[3];
  const float* gm  = (const float*)d_in[4];
  const float* bt  = (const float*)d_in[5];
  float* out = (float*)d_out;

  _Float16* Uc  = (_Float16*)d_ws;        // [512][256] f16 : 256 KB (L2-resident)
  _Float16* M2T = Uc + 512 * DIM;         // [ 64][256] f16 :  32 KB

  build_unitary<<<DIM, 256, 0, stream>>>(qw, Uc);
  build_m2<<<64, 256, 0, stream>>>(W, M2T);
  qsb_main<<<BATCH / ROWS_PER_WG, 128, 0, stream>>>(pca, Uc, M2T, b, gm, bt, out);
}
